// MultiAttention_27565100106179
// MI455X (gfx1250) — compile-verified, matrix operand order checked by reading, not run on hardware
//
#include <hip/hip_runtime.h>
#include <cstdint>
#include <cstddef>

// ---------------------------------------------------------------------------
// MI455X (gfx1250) multi-head attention, bf16 WMMA pipeline.
// GEMMs: V_WMMA_F32_16X16X32_BF16 with TDM (tensor_load_to_lds) staged,
// double-buffered B panels in LDS, register-pipelined A fragments, and
// grouped LDS fragment loads feeding back-to-back WMMAs.
// ---------------------------------------------------------------------------

typedef __attribute__((ext_vector_type(16))) __bf16 v16bf;
typedef __attribute__((ext_vector_type(8)))  __bf16 v8bf;
typedef __attribute__((ext_vector_type(8)))  float  v8f;
typedef __attribute__((ext_vector_type(4)))  unsigned int u32x4;
typedef __attribute__((ext_vector_type(4)))  int i32x4;
typedef __attribute__((ext_vector_type(8)))  int i32x8;

static constexpr int kHeads = 16;
static constexpr int kCtx   = 4096;
static constexpr int kDim   = 1024;
static constexpr int kHD    = kHeads * kDim;   // 16384

#if defined(__HIP_DEVICE_COMPILE__) && __has_builtin(__builtin_amdgcn_tensor_load_to_lds)
#define USE_TDM 1
#else
#define USE_TDM 0
#endif

__device__ __forceinline__ v8f vzero8() {
  v8f v;
#pragma unroll
  for (int i = 0; i < 8; ++i) v[i] = 0.0f;
  return v;
}

// Load a 16x32 A-fragment (or B-fragment from an NxK row-major panel).
// Per ISA 7.12.2 (16-bit A 16x32): lane L -> row (L&15),
// K chunks at k0 + (L>>4)*8 and +16 (two 16-byte contiguous loads).
__device__ __forceinline__ v16bf ldfrag(const __bf16* base, size_t row_stride,
                                        int k0) {
  const int lane = threadIdx.x & 31;
  const __bf16* p = base + (size_t)(lane & 15) * row_stride + k0 + ((lane >> 4) << 3);
  v8bf lo = *(const v8bf*)(p);
  v8bf hi = *(const v8bf*)(p + 16);
  v16bf r;
#pragma unroll
  for (int i = 0; i < 8; ++i) { r[i] = lo[i]; r[8 + i] = hi[i]; }
  return r;
}

__device__ __forceinline__ v8f wmma_bf16(v16bf a, v16bf b, v8f c) {
  return __builtin_amdgcn_wmma_f32_16x16x32_bf16(false, a, false, b,
                                                 (short)0, c, false, false);
}

#if USE_TDM
// Issue a TDM 2D tile load: tile_w elements (data_size=2B) x tile_h rows,
// row stride = stride0 elements, from gaddr into LDS byte offset lds_off.
// D# layout per CDNA5 ISA ch.8 (group0: count/lds/global/type, group1: dims).
__device__ __forceinline__ void tdm_load_2d(unsigned lds_off, const void* gaddr,
                                            unsigned td0_elems, unsigned stride0,
                                            unsigned tile_w, unsigned tile_h) {
  const unsigned long long ga = (unsigned long long)(uintptr_t)gaddr;
  u32x4 g0;
  g0[0] = 1u;                                             // count=1, user desc
  g0[1] = lds_off;                                        // lds_addr [63:32]
  g0[2] = (unsigned)(ga & 0xFFFFFFFFu);                   // global_addr lo
  g0[3] = (unsigned)((ga >> 32) & 0x1FFFFFFu) | (2u << 30);  // addr hi | type=2
  i32x8 g1;
  g1[0] = (int)(1u << 16);                                // data_size=1 (2B)
  g1[1] = (int)((td0_elems & 0xFFFFu) << 16);             // tensor_dim0 lo16
  g1[2] = (int)((td0_elems >> 16) & 0xFFFFu);             // td0 hi16 | td1 lo16=0
  g1[3] = 1 | (int)(tile_w << 16);                        // td1=65536 | tile_dim0
  g1[4] = (int)(tile_h & 0xFFFFu);                        // tile_dim1 | tile_dim2=0
  g1[5] = (int)stride0;                                   // tensor_dim0_stride lo32
  g1[6] = 0;                                              // stride hi | d1stride lo
  g1[7] = 0;
  i32x4 z4;
#pragma unroll
  for (int i = 0; i < 4; ++i) z4[i] = 0;
#if defined(__clang_major__) && __clang_major__ >= 23
  i32x8 z8;
#pragma unroll
  for (int i = 0; i < 8; ++i) z8[i] = 0;
  __builtin_amdgcn_tensor_load_to_lds(g0, g1, z4, z4, z8, 0);
#else
  __builtin_amdgcn_tensor_load_to_lds(g0, g1, z4, z4, 0);
#endif
}
#endif  // USE_TDM

// ---------------------------------------------------------------------------
// f32 -> bf16 flat conversion
// ---------------------------------------------------------------------------
__global__ void cvt_kernel(const float* __restrict__ in, __bf16* __restrict__ out,
                           size_t n) {
  size_t i = (size_t)blockIdx.x * blockDim.x + threadIdx.x;
  if (i < n) out[i] = (__bf16)in[i];
}

// ---------------------------------------------------------------------------
// f32 [R][C] -> bf16 [C][R] transpose+convert (batched over blockIdx.z).
// ---------------------------------------------------------------------------
__global__ void tcvt_kernel(const float* __restrict__ in, __bf16* __restrict__ out,
                            int R, int C) {
  __shared__ float tile[32][33];
  const size_t boff = (size_t)blockIdx.z * (size_t)R * (size_t)C;
  const int r0 = blockIdx.y * 32, c0 = blockIdx.x * 32;
  const int tx = threadIdx.x, ty = threadIdx.y;
#pragma unroll
  for (int i = 0; i < 4; ++i)
    tile[ty + 8 * i][tx] = in[boff + (size_t)(r0 + ty + 8 * i) * C + c0 + tx];
  __syncthreads();
#pragma unroll
  for (int i = 0; i < 4; ++i)
    out[boff + (size_t)(c0 + ty + 8 * i) * R + r0 + tx] = (__bf16)tile[tx][ty + 8 * i];
}

// ---------------------------------------------------------------------------
// bf16 GEMM: C = A(MxK row-major) * B, B given as BT (NxK row-major).
// Block = 256 threads = 8 waves; block tile 128(M) x 64(N); wave tile 16x64.
// The shared 64x32 B panel is staged into LDS (double-buffered) via the
// Tensor Data Mover (wave 0 issues, s_wait_tensorcnt + barrier publishes),
// A fragments are register-pipelined one K-stage ahead, and the four B
// fragments are loaded as a grouped burst feeding back-to-back WMMAs.
// flags: 1 = ReLU, 2 = bf16 out, 4 = transposed bf16 store (C[n][m]).
// ---------------------------------------------------------------------------
__global__ __launch_bounds__(256) void gemm_bf16_kernel(
    const __bf16* __restrict__ A, const __bf16* __restrict__ BT,
    const float* __restrict__ bias, void* __restrict__ Cout,
    int M, int N, int K, size_t strideB, size_t strideC, int flags) {
  __shared__ __bf16 Bs[2][64 * 32];

  const int wave = threadIdx.x >> 5;
  const int lane = threadIdx.x & 31;
  const int half = lane >> 4, idx = lane & 15;

  const int m0 = blockIdx.y * 128 + wave * 16;
  const int n0 = blockIdx.x * 64;

  const __bf16* Ab = A + (size_t)m0 * K;
  const __bf16* Bb = BT + (size_t)blockIdx.z * strideB + (size_t)n0 * K;

  const int nsteps = K >> 5;

  // ---- prologue: stage K-slice 0 into Bs[0] ------------------------------
#if USE_TDM
  if (wave == 0)
    tdm_load_2d((unsigned)(uintptr_t)&Bs[0][0], Bb, (unsigned)K, (unsigned)K,
                32u, 64u);
#else
  {
    const int t = threadIdx.x;
    v8bf v = *(const v8bf*)(Bb + (size_t)(t >> 2) * K + (t & 3) * 8);
    *(v8bf*)(&Bs[0][(t >> 2) * 32 + (t & 3) * 8]) = v;
  }
#endif
  v16bf a_c = ldfrag(Ab, (size_t)K, 0);

  v8f acc[4];
#pragma unroll
  for (int j = 0; j < 4; ++j) acc[j] = vzero8();

#if USE_TDM
  if (wave == 0) __builtin_amdgcn_s_wait_tensorcnt(0);
#endif
  __syncthreads();

  for (int kk = 0; kk < nsteps; ++kk) {
    const int cur = kk & 1, nxt = cur ^ 1;
    const bool more = (kk + 1 < nsteps);

    // issue next-stage transfers before this stage's math
#if USE_TDM
    if (more && wave == 0)
      tdm_load_2d((unsigned)(uintptr_t)&Bs[nxt][0], Bb + (size_t)(kk + 1) * 32,
                  (unsigned)K, (unsigned)K, 32u, 64u);
#else
    v8bf breg;
    if (more) {
      const int t = threadIdx.x;
      breg = *(const v8bf*)(Bb + (size_t)(t >> 2) * K + (size_t)(kk + 1) * 32 +
                            (t & 3) * 8);
    }
#endif
    v16bf a_n;
    if (more) a_n = ldfrag(Ab, (size_t)K, (kk + 1) * 32);
    if (kk + 4 < nsteps)
      __builtin_prefetch(Ab + (size_t)idx * K + (size_t)(kk + 4) * 32, 0, 0);

    // math for this stage: grouped LDS fragment loads, then WMMA burst
    v16bf bfr[4];
#pragma unroll
    for (int j = 0; j < 4; ++j)
      bfr[j] = ldfrag(&Bs[cur][j * 16 * 32], 32, 0);
#pragma unroll
    for (int j = 0; j < 4; ++j)
      acc[j] = wmma_bf16(a_c, bfr[j], acc[j]);

    if (more) {
      a_c = a_n;
#if USE_TDM
      if (wave == 0) __builtin_amdgcn_s_wait_tensorcnt(0);
#else
      {
        const int t = threadIdx.x;
        *(v8bf*)(&Bs[nxt][(t >> 2) * 32 + (t & 3) * 8]) = breg;
      }
#endif
      __syncthreads();
    }
  }

  // ---- epilogue ----------------------------------------------------------
  const size_t coff = (size_t)blockIdx.z * strideC;
#pragma unroll
  for (int j = 0; j < 4; ++j) {
    const int n = n0 + j * 16 + idx;
    const float bv = bias ? bias[n] : 0.0f;
#pragma unroll
    for (int r = 0; r < 8; ++r) {
      const int m = m0 + r + half * 8;   // C layout: M = r + (lane>>4)*8
      float v = acc[j][r] + bv;
      if (flags & 1) v = fmaxf(v, 0.0f);
      if (flags & 4)
        ((__bf16*)Cout)[coff + (size_t)n * M + m] = (__bf16)v;
      else if (flags & 2)
        ((__bf16*)Cout)[coff + (size_t)m * N + n] = (__bf16)v;
      else
        ((float*)Cout)[coff + (size_t)m * N + n] = v;
    }
  }
}

// ---------------------------------------------------------------------------
// Causal flash attention.  Grid: (S/16, H).  Block: 256 threads = 8 waves.
// Per block: one head, 16 query rows; key blocks of 256:
//   - each wave: scores for 32 keys (2 N-tiles), register-pipelined K loop
//   - thread-per-row streaming softmax -> bf16 P in LDS
//   - each wave: Y over its 128-column slice (8 f32 tiles), grouped B loads
// Q, Kmat: bf16 [H][S][D].  VT: bf16 [H][D][S].  Ycat: bf16 [S][H*D].
// ---------------------------------------------------------------------------
__global__ __launch_bounds__(256) void attn_kernel(
    const __bf16* __restrict__ Q, const __bf16* __restrict__ Kmat,
    const __bf16* __restrict__ VT, __bf16* __restrict__ Ycat) {
  __shared__ float  Sbuf[16][256];
  __shared__ __bf16 Pbuf[16][256];
  __shared__ float  mrow[16], lrow[16], arow[16];

  const int h  = blockIdx.y;
  const int m0 = blockIdx.x * 16;
  const int wave = threadIdx.x >> 5;
  const int lane = threadIdx.x & 31;
  const int half = lane >> 4, idx = lane & 15;

  if (threadIdx.x < 16) { mrow[threadIdx.x] = -3.0e38f; lrow[threadIdx.x] = 0.0f; }
  __syncthreads();

  const __bf16* Qh = Q    + ((size_t)h * kCtx + m0) * kDim;
  const __bf16* Kh = Kmat + (size_t)h * kCtx * kDim;
  const __bf16* Vh = VT   + (size_t)h * kDim * kCtx;

  v8f acc[8];
#pragma unroll
  for (int t = 0; t < 8; ++t) acc[t] = vzero8();

  const int nkb = m0 / 256 + 1;  // causal: only key blocks with keys <= m0+15
  for (int kb = 0; kb < nkb; ++kb) {
    const int keys0 = kb * 256;

    // ---- scores: wave handles keys [keys0+wave*32, +32), pipelined -------
    v8f s0 = vzero8(), s1 = vzero8();
    const __bf16* Kb0 = Kh + (size_t)(keys0 + wave * 32) * kDim;
    const __bf16* Kb1 = Kb0 + (size_t)16 * kDim;
    v16bf a_c  = ldfrag(Qh,  kDim, 0);
    v16bf b0_c = ldfrag(Kb0, kDim, 0);
    v16bf b1_c = ldfrag(Kb1, kDim, 0);
    for (int k0 = 0; k0 < kDim - 32; k0 += 32) {
      v16bf a_n  = ldfrag(Qh,  kDim, k0 + 32);
      v16bf b0_n = ldfrag(Kb0, kDim, k0 + 32);
      v16bf b1_n = ldfrag(Kb1, kDim, k0 + 32);
      s0 = wmma_bf16(a_c, b0_c, s0);
      s1 = wmma_bf16(a_c, b1_c, s1);
      a_c = a_n; b0_c = b0_n; b1_c = b1_n;
    }
    s0 = wmma_bf16(a_c, b0_c, s0);
    s1 = wmma_bf16(a_c, b1_c, s1);

#pragma unroll
    for (int r = 0; r < 8; ++r) {
      Sbuf[r + half * 8][wave * 32 + idx]      = s0[r];
      Sbuf[r + half * 8][wave * 32 + 16 + idx] = s1[r];
    }
    __syncthreads();

    // ---- streaming softmax: one thread per query row ---------------------
    if (threadIdx.x < 16) {
      const int row = threadIdx.x;
      const int q   = m0 + row;
      const float mold = mrow[row];
      float bm = -3.0e38f;
      for (int j = 0; j < 256; ++j) {
        const int key = keys0 + j;
        float s = (key <= q) ? Sbuf[row][j] * 0.03125f : -3.0e38f;  // 1/sqrt(1024)
        Sbuf[row][j] = s;
        bm = fmaxf(bm, s);
      }
      const float mnew  = fmaxf(mold, bm);
      const float alpha = __expf(mold - mnew);
      float sum = 0.0f;
      for (int j = 0; j < 256; ++j) {
        const float p = __expf(Sbuf[row][j] - mnew);
        Pbuf[row][j] = (__bf16)p;
        sum += p;
      }
      lrow[row] = alpha * lrow[row] + sum;
      mrow[row] = mnew;
      arow[row] = alpha;
    }
    __syncthreads();

    // ---- rescale running Y, accumulate P @ V (grouped loads) -------------
    float al[8];
#pragma unroll
    for (int r = 0; r < 8; ++r) al[r] = arow[r + half * 8];
#pragma unroll
    for (int t = 0; t < 8; ++t)
#pragma unroll
      for (int r = 0; r < 8; ++r) acc[t][r] *= al[r];

    const int c0 = wave * 128;
    for (int kk = 0; kk < 256; kk += 32) {
      v16bf a = ldfrag((const __bf16*)&Pbuf[0][0], 256, kk);
      v16bf bfr[8];
#pragma unroll
      for (int t = 0; t < 8; ++t)
        bfr[t] = ldfrag(Vh + (size_t)(c0 + t * 16) * kCtx, kCtx, keys0 + kk);
#pragma unroll
      for (int t = 0; t < 8; ++t)
        acc[t] = wmma_bf16(a, bfr[t], acc[t]);
    }
    __syncthreads();  // protect Sbuf/Pbuf before next key block
  }

  // ---- normalize, store into concatenated-head layout --------------------
  float inv[8];
#pragma unroll
  for (int r = 0; r < 8; ++r) inv[r] = 1.0f / lrow[r + half * 8];
  const size_t colbase = (size_t)h * kDim + wave * 128;
#pragma unroll
  for (int t = 0; t < 8; ++t)
#pragma unroll
    for (int r = 0; r < 8; ++r)
      Ycat[(size_t)(m0 + r + half * 8) * kHD + colbase + t * 16 + idx] =
          (__bf16)(acc[t][r] * inv[r]);
}

// ---------------------------------------------------------------------------
// Host launch
// ---------------------------------------------------------------------------
extern "C" void kernel_launch(void* const* d_in, const int* in_sizes, int n_in,
                              void* d_out, int out_size, void* d_ws, size_t ws_size,
                              hipStream_t stream) {
  (void)in_sizes; (void)n_in; (void)out_size; (void)ws_size;
  const int H = kHeads, S = kCtx, D = kDim;

  const float* X  = (const float*)d_in[0];
  const float* Wq = (const float*)d_in[1];
  const float* Wk = (const float*)d_in[2];
  const float* Wv = (const float*)d_in[3];
  const float* Wp = (const float*)d_in[4];
  const float* bp = (const float*)d_in[5];
  const float* Wl = (const float*)d_in[6];
  const float* bl = (const float*)d_in[7];

  char* ws = (char*)d_ws;
  size_t off = 0;
  auto alloc = [&](size_t bytes) -> void* {
    void* p = ws + off;
    off = (off + bytes + 255) & ~(size_t)255;
    return p;
  };
  __bf16* Xh   = (__bf16*)alloc((size_t)S * D * 2);
  __bf16* WqT  = (__bf16*)alloc((size_t)H * D * D * 2);
  __bf16* WkT  = (__bf16*)alloc((size_t)H * D * D * 2);
  __bf16* WvT  = (__bf16*)alloc((size_t)H * D * D * 2);
  __bf16* WpT  = (__bf16*)alloc((size_t)kHD * D * 2);
  __bf16* WlT  = (__bf16*)alloc((size_t)D * D * 2);
  __bf16* Qb   = (__bf16*)alloc((size_t)H * S * D * 2);
  __bf16* Kb   = (__bf16*)alloc((size_t)H * S * D * 2);
  __bf16* VTb  = (__bf16*)alloc((size_t)H * D * S * 2);
  __bf16* Ycat = (__bf16*)alloc((size_t)S * kHD * 2);
  __bf16* H1   = (__bf16*)alloc((size_t)S * D * 2);

  // 1) precision conversion / weight transposes
  cvt_kernel<<<(S * D) / 256, 256, 0, stream>>>(X, Xh, (size_t)S * D);
  dim3 tb(32, 8);
  tcvt_kernel<<<dim3(D / 32, D / 32, H), tb, 0, stream>>>(Wq, WqT, D, D);
  tcvt_kernel<<<dim3(D / 32, D / 32, H), tb, 0, stream>>>(Wk, WkT, D, D);
  tcvt_kernel<<<dim3(D / 32, D / 32, H), tb, 0, stream>>>(Wv, WvT, D, D);
  tcvt_kernel<<<dim3(D / 32, kHD / 32, 1), tb, 0, stream>>>(Wp, WpT, kHD, D);
  tcvt_kernel<<<dim3(D / 32, D / 32, 1), tb, 0, stream>>>(Wl, WlT, D, D);

  // 2) Q/K/V projections (batched over heads); V stored transposed [H][D][S]
  dim3 gproj(D / 64, S / 128, H);
  gemm_bf16_kernel<<<gproj, 256, 0, stream>>>(Xh, WqT, nullptr, Qb, S, D, D,
      (size_t)D * D, (size_t)S * D, /*bf16 out*/ 2);
  gemm_bf16_kernel<<<gproj, 256, 0, stream>>>(Xh, WkT, nullptr, Kb, S, D, D,
      (size_t)D * D, (size_t)S * D, 2);
  gemm_bf16_kernel<<<gproj, 256, 0, stream>>>(Xh, WvT, nullptr, VTb, S, D, D,
      (size_t)D * D, (size_t)D * S, /*transposed store*/ 4);

  // 3) causal flash attention -> Ycat [S][H*D]
  attn_kernel<<<dim3(S / 16, H), 256, 0, stream>>>(Qb, Kb, VTb, Ycat);

  // 4) relu(Ycat @ Wp + bp) -> H1 (bf16), then H1 @ Wl + bl -> f32 out
  gemm_bf16_kernel<<<dim3(D / 64, S / 128, 1), 256, 0, stream>>>(
      Ycat, WpT, bp, H1, S, D, kHD, 0, 0, /*relu|bf16*/ 3);
  gemm_bf16_kernel<<<dim3(D / 64, S / 128, 1), 256, 0, stream>>>(
      H1, WlT, bl, d_out, S, D, D, 0, 0, /*f32 out*/ 0);
}
